// TreeLSTM_82875688944085
// MI455X (gfx1250) — compile-verified, matrix operand order checked by reading, not run on hardware
//
#include <hip/hip_runtime.h>

// ---------------------------------------------------------------------------
// Types for CDNA5 WMMA (wave32): D(16x16 f32) = A(16x32 bf16) x B(32x16 bf16) + C
// ---------------------------------------------------------------------------
typedef __attribute__((ext_vector_type(16))) __bf16 v16bf;
typedef __attribute__((ext_vector_type(8)))  __bf16 v8bf;
typedef __attribute__((ext_vector_type(8)))  float  v8f;

#define HD 256
#define MT 4                    // M-subtiles per block (M = 64 rows)
#define BM (MT * 16)

__device__ __forceinline__ unsigned short f2bf(float x) {
    union { float f; unsigned u; } c; c.f = x;
    unsigned u = c.u;
    return (unsigned short)((u + 0x7fffu + ((u >> 16) & 1u)) >> 16);  // RNE
}
__device__ __forceinline__ float bf2f(unsigned short b) {
    union { unsigned u; float f; } c; c.u = ((unsigned)b) << 16;
    return c.f;
}
__device__ __forceinline__ float sigm(float x) { return 1.0f / (1.0f + __expf(-x)); }

// Load one 16x32 bf16 fragment operand (A or B) for this lane.
// ISA 7.12.2 16-bit layout: within a 32-wide K slice, lanes 0-15 hold
// K[0..7],[16..23]; lanes 16-31 hold K[8..15],[24..31]. Caller bakes the
// per-lane +8 (half*8) into `p`, so we read p[0..7] and p[16..23].
__device__ __forceinline__ v16bf ldfrag(const unsigned short* p) {
    v8bf lo = *reinterpret_cast<const v8bf*>(p);
    v8bf hi = *reinterpret_cast<const v8bf*>(p + 16);
    return __builtin_shufflevector(lo, hi, 0, 1, 2, 3, 4, 5, 6, 7,
                                           8, 9, 10, 11, 12, 13, 14, 15);
}

#define WMMA_BF16(A, B, C) \
    __builtin_amdgcn_wmma_f32_16x16x32_bf16(false, (A), false, (B), (short)0, (C), false, false)

// ---------------------------------------------------------------------------
// Weight f32 -> bf16 conversion
// ---------------------------------------------------------------------------
__global__ void cvt_kernel(const float* __restrict__ s, unsigned short* __restrict__ d, int n) {
    int i = blockIdx.x * blockDim.x + threadIdx.x;
    if (i < n) d[i] = f2bf(s[i]);
}

// ---------------------------------------------------------------------------
// Leaf stage: emb = embed[word_ids]; i,o,u = act(emb @ W^T + b); c=i*u; h=o*tanh(c)
// Block = 64 rows, 512 threads (16 waves). Wave w owns N-tile w (16 cols) for
// all 4 M-subtiles; B fragments are loaded once per k and reused 4x.
// ---------------------------------------------------------------------------
__global__ __launch_bounds__(512)
void leaf_kernel(const int* __restrict__ wid, const float* __restrict__ embed,
                 const unsigned short* __restrict__ wiw, const float* __restrict__ wib,
                 const unsigned short* __restrict__ wow, const float* __restrict__ wob,
                 const unsigned short* __restrict__ wuw, const float* __restrict__ wub,
                 unsigned short* __restrict__ h_out, float* __restrict__ c_out)
{
    __shared__ unsigned short Abuf[BM * HD];   // 64 x 256 bf16 = 32 KB
    __shared__ int rows[BM];

    const int t  = threadIdx.x;
    const int r0 = blockIdx.x * BM;

    if (t < BM) rows[t] = wid[r0 + t];
    __syncthreads();

    // Gather + convert: each thread moves 32 consecutive floats of one row.
    {
        const int base = t * 32;
        const int row = base >> 8, col = base & 255;
        const float* src = embed + (size_t)rows[row] * HD + col;
        unsigned short* dst = Abuf + row * HD + col;
#pragma unroll
        for (int i = 0; i < 8; ++i) {
            float4 f = reinterpret_cast<const float4*>(src)[i];
            unsigned long long pk =
                (unsigned long long)f2bf(f.x)
              | ((unsigned long long)f2bf(f.y) << 16)
              | ((unsigned long long)f2bf(f.z) << 32)
              | ((unsigned long long)f2bf(f.w) << 48);
            *reinterpret_cast<unsigned long long*>(dst + i * 4) = pk;
        }
    }
    __syncthreads();

    const int wave = t >> 5, lane = t & 31;
    const int half = lane >> 4, m = lane & 15;
    const int col  = wave * 16 + m;            // this lane's N index (also B row)

    v8f ai[MT], ao[MT], au[MT];
#pragma unroll
    for (int mt = 0; mt < MT; ++mt) { ai[mt] = {}; ao[mt] = {}; au[mt] = {}; }

#pragma unroll
    for (int k = 0; k < 8; ++k) {              // K = 256 = 8 x 32
        const int off = k * 32 + half * 8;
        v16bf bi = ldfrag(wiw + (size_t)col * HD + off);
        v16bf bo = ldfrag(wow + (size_t)col * HD + off);
        v16bf bu = ldfrag(wuw + (size_t)col * HD + off);
#pragma unroll
        for (int mt = 0; mt < MT; ++mt) {
            v16bf a = ldfrag(Abuf + (mt * 16 + m) * HD + off);
            ai[mt] = WMMA_BF16(a, bi, ai[mt]);
            ao[mt] = WMMA_BF16(a, bo, ao[mt]);
            au[mt] = WMMA_BF16(a, bu, au[mt]);
        }
    }

    const float bI = wib[col], bO = wob[col], bU = wub[col];
#pragma unroll
    for (int mt = 0; mt < MT; ++mt) {
#pragma unroll
        for (int v = 0; v < 8; ++v) {          // C layout: M = v + half*8, N = col
            const int row = r0 + mt * 16 + v + half * 8;
            float iv = sigm(ai[mt][v] + bI);
            float ov = sigm(ao[mt][v] + bO);
            float uv = tanhf(au[mt][v] + bU);
            float cc = iv * uv;
            c_out[(size_t)row * HD + col] = cc;
            h_out[(size_t)row * HD + col] = f2bf(ov * tanhf(cc));
        }
    }
}

// ---------------------------------------------------------------------------
// One tree level: for parent p, e=[h(2p)|h(2p+1)] (512);
// i,o,u = act(e @ U^T + b) [K=512]; f1 = sig(h1 @ Uf1^T); f2 = sig(h2 @ Uf2^T);
// c = i*u + f1*c1 + f2*c2 ; h = o*tanh(c).
// Block = 64 parents, 512 threads; B fragments reused across 4 M-subtiles.
// ---------------------------------------------------------------------------
__global__ __launch_bounds__(512)
void node_kernel(const unsigned short* __restrict__ h_in, const float* __restrict__ c_in,
                 unsigned short* __restrict__ h_out, float* __restrict__ c_out,
                 const unsigned short* __restrict__ uiw, const float* __restrict__ uib,
                 const unsigned short* __restrict__ uow, const float* __restrict__ uob,
                 const unsigned short* __restrict__ uuw, const float* __restrict__ uub,
                 const unsigned short* __restrict__ uf1w, const float* __restrict__ uf1b,
                 const unsigned short* __restrict__ uf2w, const float* __restrict__ uf2b,
                 int n_par)
{
    __shared__ unsigned short Ebuf[BM * 512];  // 64 parents x 512 bf16 = 64 KB

    const int t  = threadIdx.x;
    const int p0 = blockIdx.x * BM;

    // Build e = [h1 | h2] in LDS; each thread copies 64 bf16 as 8 x 16B chunks.
    {
        const int base = t * 64;
        const int p = base >> 9;
        uint4* dst = reinterpret_cast<uint4*>(Ebuf + base);
        const bool valid = (p0 + p) < n_par;
#pragma unroll
        for (int j = 0; j < 8; ++j) {
            const int d = (base & 511) + j * 8;          // 8-aligned, never splits a chunk
            if (valid) {
                const int child = 2 * (p0 + p) + (d >= 256 ? 1 : 0);
                dst[j] = *reinterpret_cast<const uint4*>(h_in + (size_t)child * HD + (d & 255));
            } else {
                dst[j] = uint4{0u, 0u, 0u, 0u};
            }
        }
    }

    const int wave = t >> 5, lane = t & 31;
    const int half = lane >> 4, m = lane & 15;
    const int col  = wave * 16 + m;

    // Warm the epilogue's c1/c2 cachelines while the GEMM runs.
#pragma unroll
    for (int mt = 0; mt < MT; ++mt) {
        const int p = p0 + mt * 16 + half * 8;
        if (p < n_par) {
            __builtin_prefetch(c_in + (size_t)(2 * p) * HD + col, 0, 3);
            __builtin_prefetch(c_in + (size_t)(2 * p + 1) * HD + col, 0, 3);
        }
    }
    __syncthreads();

    v8f ai[MT], ao[MT], au[MT], af1[MT], af2[MT];
#pragma unroll
    for (int mt = 0; mt < MT; ++mt) { ai[mt] = {}; ao[mt] = {}; au[mt] = {}; af1[mt] = {}; af2[mt] = {}; }

    // K 0..255 : e-half h1 ; f-gate uses Uf1
#pragma unroll
    for (int k = 0; k < 8; ++k) {
        const int off = k * 32 + half * 8;
        v16bf bi = ldfrag(uiw  + (size_t)col * 512 + off);
        v16bf bo = ldfrag(uow  + (size_t)col * 512 + off);
        v16bf bu = ldfrag(uuw  + (size_t)col * 512 + off);
        v16bf bf = ldfrag(uf1w + (size_t)col * HD  + off);
#pragma unroll
        for (int mt = 0; mt < MT; ++mt) {
            v16bf a = ldfrag(Ebuf + (mt * 16 + m) * 512 + off);
            ai[mt]  = WMMA_BF16(a, bi, ai[mt]);
            ao[mt]  = WMMA_BF16(a, bo, ao[mt]);
            au[mt]  = WMMA_BF16(a, bu, au[mt]);
            af1[mt] = WMMA_BF16(a, bf, af1[mt]);
        }
    }
    // K 256..511 : e-half h2 ; f-gate uses Uf2 (K index re-based by -256)
#pragma unroll
    for (int k = 8; k < 16; ++k) {
        const int off = k * 32 + half * 8;
        v16bf bi = ldfrag(uiw  + (size_t)col * 512 + off);
        v16bf bo = ldfrag(uow  + (size_t)col * 512 + off);
        v16bf bu = ldfrag(uuw  + (size_t)col * 512 + off);
        v16bf bf = ldfrag(uf2w + (size_t)col * HD  + (off - 256));
#pragma unroll
        for (int mt = 0; mt < MT; ++mt) {
            v16bf a = ldfrag(Ebuf + (mt * 16 + m) * 512 + off);
            ai[mt]  = WMMA_BF16(a, bi, ai[mt]);
            ao[mt]  = WMMA_BF16(a, bo, ao[mt]);
            au[mt]  = WMMA_BF16(a, bu, au[mt]);
            af2[mt] = WMMA_BF16(a, bf, af2[mt]);
        }
    }

    const float bI = uib[col], bO = uob[col], bU = uub[col];
    const float bF1 = uf1b[col], bF2 = uf2b[col];
#pragma unroll
    for (int mt = 0; mt < MT; ++mt) {
#pragma unroll
        for (int v = 0; v < 8; ++v) {
            const int p = p0 + mt * 16 + v + half * 8;
            if (p < n_par) {
                float iv  = sigm(ai[mt][v]  + bI);
                float ov  = sigm(ao[mt][v]  + bO);
                float uv  = tanhf(au[mt][v] + bU);
                float f1v = sigm(af1[mt][v] + bF1);
                float f2v = sigm(af2[mt][v] + bF2);
                float c1 = c_in[(size_t)(2 * p)     * HD + col];
                float c2 = c_in[(size_t)(2 * p + 1) * HD + col];
                float cc = iv * uv + f1v * c1 + f2v * c2;
                c_out[(size_t)p * HD + col] = cc;
                h_out[(size_t)p * HD + col] = f2bf(ov * tanhf(cc));
            }
        }
    }
}

// ---------------------------------------------------------------------------
// Head: logits = relu(h @ wf^T + wf_b) @ out_w^T + out_b  (h is 1x256)
// ---------------------------------------------------------------------------
__global__ __launch_bounds__(256)
void final_kernel(const unsigned short* __restrict__ h,
                  const float* __restrict__ wfw, const float* __restrict__ wfb,
                  const float* __restrict__ outw, const float* __restrict__ outb,
                  float* __restrict__ out)
{
    __shared__ float hrow[HD];
    __shared__ float hf[HD];
    const int t = threadIdx.x;
    hrow[t] = bf2f(h[t]);
    __syncthreads();
    float acc = wfb[t];
    const float* wr = wfw + (size_t)t * HD;
#pragma unroll 8
    for (int k = 0; k < HD; ++k) acc += hrow[k] * wr[k];
    hf[t] = fmaxf(acc, 0.0f);
    __syncthreads();
    if (t < 5) {
        float s = outb[t];
        const float* ow = outw + (size_t)t * HD;
#pragma unroll 8
        for (int k = 0; k < HD; ++k) s += hf[k] * ow[k];
        out[t] = s;
    }
}

// ---------------------------------------------------------------------------
// Launch
// ---------------------------------------------------------------------------
extern "C" void kernel_launch(void* const* d_in, const int* in_sizes, int n_in,
                              void* d_out, int out_size, void* d_ws, size_t ws_size,
                              hipStream_t stream)
{
    (void)in_sizes; (void)n_in; (void)out_size; (void)ws_size;

    const int*   word_ids = (const int*)  d_in[0];
    const float* embed    = (const float*)d_in[1];
    const float* wi_w = (const float*)d_in[2];  const float* wi_b = (const float*)d_in[3];
    const float* wo_w = (const float*)d_in[4];  const float* wo_b = (const float*)d_in[5];
    const float* wu_w = (const float*)d_in[6];  const float* wu_b = (const float*)d_in[7];
    const float* ui_w = (const float*)d_in[8];  const float* ui_b = (const float*)d_in[9];
    const float* uo_w = (const float*)d_in[10]; const float* uo_b = (const float*)d_in[11];
    const float* uu_w = (const float*)d_in[12]; const float* uu_b = (const float*)d_in[13];
    const float* uf1_w = (const float*)d_in[14]; const float* uf1_b = (const float*)d_in[15];
    const float* uf2_w = (const float*)d_in[16]; const float* uf2_b = (const float*)d_in[17];
    const float* wf_w = (const float*)d_in[18]; const float* wf_b = (const float*)d_in[19];
    const float* out_w = (const float*)d_in[20]; const float* out_b = (const float*)d_in[21];

    const int NL = 65536;
    size_t off = 0;
    auto carve = [&](size_t bytes) -> char* {
        char* p = (char*)d_ws + off;
        off += (bytes + 255) & ~(size_t)255;
        return p;
    };
    unsigned short* wiw  = (unsigned short*)carve((size_t)HD * HD * 2);
    unsigned short* wow  = (unsigned short*)carve((size_t)HD * HD * 2);
    unsigned short* wuw  = (unsigned short*)carve((size_t)HD * HD * 2);
    unsigned short* uiw  = (unsigned short*)carve((size_t)HD * 2 * HD * 2);
    unsigned short* uow  = (unsigned short*)carve((size_t)HD * 2 * HD * 2);
    unsigned short* uuw  = (unsigned short*)carve((size_t)HD * 2 * HD * 2);
    unsigned short* uf1w = (unsigned short*)carve((size_t)HD * HD * 2);
    unsigned short* uf2w = (unsigned short*)carve((size_t)HD * HD * 2);
    unsigned short* hA   = (unsigned short*)carve((size_t)NL * HD * 2);
    unsigned short* hB   = (unsigned short*)carve((size_t)(NL / 2) * HD * 2);
    float*          cA   = (float*)carve((size_t)NL * HD * 4);
    float*          cB   = (float*)carve((size_t)(NL / 2) * HD * 4);

    // 1) weights -> bf16
    const int n1 = HD * HD, n2 = HD * 2 * HD;
    cvt_kernel<<<n1 / 256, 256, 0, stream>>>(wi_w,  wiw,  n1);
    cvt_kernel<<<n1 / 256, 256, 0, stream>>>(wo_w,  wow,  n1);
    cvt_kernel<<<n1 / 256, 256, 0, stream>>>(wu_w,  wuw,  n1);
    cvt_kernel<<<n2 / 256, 256, 0, stream>>>(ui_w,  uiw,  n2);
    cvt_kernel<<<n2 / 256, 256, 0, stream>>>(uo_w,  uow,  n2);
    cvt_kernel<<<n2 / 256, 256, 0, stream>>>(uu_w,  uuw,  n2);
    cvt_kernel<<<n1 / 256, 256, 0, stream>>>(uf1_w, uf1w, n1);
    cvt_kernel<<<n1 / 256, 256, 0, stream>>>(uf2_w, uf2w, n1);

    // 2) leaf stage -> hA, cA
    leaf_kernel<<<NL / BM, 512, 0, stream>>>(word_ids, embed,
                                             wiw, wi_b, wow, wo_b, wuw, wu_b,
                                             hA, cA);

    // 3) 16 tree levels, ping-pong A<->B
    unsigned short* h_cur = hA; float* c_cur = cA;
    unsigned short* h_nxt = hB; float* c_nxt = cB;
    int n_par = NL / 2;
    for (int l = 0; l < 16; ++l) {
        int blocks = (n_par + BM - 1) / BM;
        node_kernel<<<blocks, 512, 0, stream>>>(h_cur, c_cur, h_nxt, c_nxt,
                                                uiw, ui_b, uow, uo_b, uuw, uu_b,
                                                uf1w, uf1_b, uf2w, uf2_b, n_par);
        unsigned short* th = h_cur; h_cur = h_nxt; h_nxt = th;
        float* tc = c_cur; c_cur = c_nxt; c_nxt = tc;
        n_par >>= 1;
    }

    // 4) head (root h is h_cur row 0)
    final_kernel<<<1, 256, 0, stream>>>(h_cur, wf_w, wf_b, out_w, out_b, (float*)d_out);
}
